// Encoder_base_87393994539038
// MI455X (gfx1250) — compile-verified, matrix operand order checked by reading, not run on hardware
//
#include <hip/hip_runtime.h>
#include <hip/hip_bf16.h>
#include <math.h>

typedef __attribute__((ext_vector_type(16))) _Float16 v16h;
typedef __attribute__((ext_vector_type(8)))  float    v8f;

// ---------------------------------------------------------------------------
// Utility kernels
// ---------------------------------------------------------------------------
__global__ void zero_f(float* __restrict__ p, long long n) {
  long long i = (long long)blockIdx.x * blockDim.x + threadIdx.x;
  if (i < n) p[i] = 0.f;
}

// deg[row] += 1 for each edge (edge[0:E] = rows)
__global__ void deg_acc(const int* __restrict__ edge, int E, float* __restrict__ deg) {
  int i = blockIdx.x * blockDim.x + threadIdx.x;
  if (i < E) atomicAdd(&deg[edge[i]], 1.0f);
}

// deg -> dis = (deg>0) ? rsqrt(max(deg,1)) : 0   (in place)
__global__ void deg_to_dis(float* __restrict__ deg, int N) {
  int i = blockIdx.x * blockDim.x + threadIdx.x;
  if (i < N) {
    float d = deg[i];
    deg[i] = (d > 0.f) ? rsqrtf(fmaxf(d, 1.f)) : 0.f;
  }
}

// out[b,row,c] += x[b,col,c] * (-dis[row]*dis[col])   (scatter over edges)
__global__ void prop_k(const float* __restrict__ x, const int* __restrict__ edge,
                       const float* __restrict__ dis, float* __restrict__ out,
                       int B, int N, int C, int E) {
  int t = blockIdx.x * blockDim.x + threadIdx.x;
  if (t >= E * C) return;
  int e = t / C, c = t % C;
  int r = edge[e];
  int cl = edge[E + e];
  float w = -dis[r] * dis[cl];
  for (int b = 0; b < B; ++b) {
    atomicAdd(&out[((long long)b * N + r) * C + c],
              x[((long long)b * N + cl) * C + c] * w);
  }
}

// out[b,j,h] = (opt tanh)( bias[h] + sum_c x*W0 + Tx1*W1 + (2*Txt - x)*W2 )
// evaluated only at nodes n = idx[j] (idx==nullptr -> identity)
__global__ void cheb_combine(const float* __restrict__ x, const float* __restrict__ tx1,
                             const float* __restrict__ txt, const float* __restrict__ W,
                             const float* __restrict__ bias, const int* __restrict__ idx,
                             float* __restrict__ out, int B, int N, int C, int H,
                             int Nout, int do_tanh) {
  int t = blockIdx.x * blockDim.x + threadIdx.x;
  if (t >= B * Nout * H) return;
  int h = t % H;
  int j = (t / H) % Nout;
  int b = t / (H * Nout);
  int n = idx ? idx[j] : j;
  const float* xr = x   + ((long long)b * N + n) * C;
  const float* t1 = tx1 + ((long long)b * N + n) * C;
  const float* tt = txt + ((long long)b * N + n) * C;
  float acc = bias[h];
  for (int c = 0; c < C; ++c) {
    float x0 = xr[c];
    float x1 = t1[c];
    float x2 = 2.f * tt[c] - x0;
    acc += x0 * W[c * H + h] + x1 * W[(C + c) * H + h] + x2 * W[(2 * C + c) * H + h];
  }
  if (do_tanh) acc = tanhf(acc);
  out[((long long)b * Nout + j) * H + h] = acc;
}

// One-shot split of the small activation matrix X into hi/lo f16 planes.
// X is L2-resident; this removes all A-side conversion VALU from the GEMM loop.
__global__ void split_f16(const float* __restrict__ X, _Float16* __restrict__ hi,
                          _Float16* __restrict__ lo, long long n) {
  long long i = (long long)blockIdx.x * blockDim.x + threadIdx.x;
  if (i < n) {
    float v = X[i];
    _Float16 h = (_Float16)v;
    hi[i] = h;
    lo[i] = (_Float16)(v - (float)h);
  }
}

// ---------------------------------------------------------------------------
// Split-precision WMMA GEMM: out[32,N] = X[32,K] @ W[K,N]
//   X pre-split into hi/lo f16 (row-major [32,K]); W f32 streamed + split
//   in-loop. Accumulate xh*wh + xl*wh + xh*wl in f32 WMMA acc (~22 mantissa
//   bits ~= f32). One wave owns a 32x16 output tile, loops K by 32.
// A layout (contiguous k order used by this code): lane holds row m=lane%16, K = k + 16*(lane/16) .. +15
//   -> direct 32B v16h loads from the pre-split planes.
// B layout: lane holds col n=lane%16, same K range, contiguous per lane.
// C/D: VGPR r -> row r + 8*(lane/16), col = lane%16.
// W is streamed exactly once -> non-temporal loads keep it out of L2's way.
// ---------------------------------------------------------------------------
__global__ void gemm32_wmma(const _Float16* __restrict__ Ah, const _Float16* __restrict__ Al,
                            const float* __restrict__ W, float* __restrict__ out,
                            int K, int N) {
  int wave = blockIdx.x * (blockDim.x >> 5) + (threadIdx.x >> 5);
  int lane = threadIdx.x & 31;
  int n0 = wave * 16;
  if (n0 >= N) return;                 // wave-uniform: EXEC stays all-ones
  int col   = lane & 15;
  int khalf = (lane >> 4) << 4;        // 0 or 16
  v8f acc0 = {};
  v8f acc1 = {};
  long long r0 = (long long)col * K + khalf;        // A row m = col
  long long r1 = (long long)(col + 16) * K + khalf; // A row m = col + 16
  for (int k = 0; k < K; k += 32) {
    const float* wp = W + (long long)(k + khalf) * N + n0 + col;
    if (k + 32 < K) {
      // pull next K-slab of the streaming weight matrix -> global_prefetch_b8
      __builtin_prefetch(wp + (long long)32 * N, 0, 1);
    }
    v16h a0h = *(const v16h*)(Ah + r0 + k);
    v16h a0l = *(const v16h*)(Al + r0 + k);
    v16h a1h = *(const v16h*)(Ah + r1 + k);
    v16h a1l = *(const v16h*)(Al + r1 + k);
    v16h bh, bl;
#pragma unroll
    for (int i = 0; i < 16; ++i) {
      float w = __builtin_nontemporal_load(wp + (long long)i * N);
      _Float16 hw = (_Float16)w;
      bh[i] = hw;
      bl[i] = (_Float16)(w - (float)hw);
    }
    acc0 = __builtin_amdgcn_wmma_f32_16x16x32_f16(false, a0h, false, bh, (short)0,
                                                  acc0, false, false);
    acc0 = __builtin_amdgcn_wmma_f32_16x16x32_f16(false, a0l, false, bh, (short)0,
                                                  acc0, false, false);
    acc0 = __builtin_amdgcn_wmma_f32_16x16x32_f16(false, a0h, false, bl, (short)0,
                                                  acc0, false, false);
    acc1 = __builtin_amdgcn_wmma_f32_16x16x32_f16(false, a1h, false, bh, (short)0,
                                                  acc1, false, false);
    acc1 = __builtin_amdgcn_wmma_f32_16x16x32_f16(false, a1l, false, bh, (short)0,
                                                  acc1, false, false);
    acc1 = __builtin_amdgcn_wmma_f32_16x16x32_f16(false, a1h, false, bl, (short)0,
                                                  acc1, false, false);
  }
  int rbase = (lane >> 4) << 3;        // 0 or 8
#pragma unroll
  for (int r = 0; r < 8; ++r) {
    out[(long long)(rbase + r) * N + n0 + col]      = acc0[r];
    out[(long long)(16 + rbase + r) * N + n0 + col] = acc1[r];
  }
}

// BatchNorm over batch dim (B=32 rows), optional affine+ReLU
__global__ void bn_relu(const float* __restrict__ H, const float* __restrict__ g,
                        const float* __restrict__ be, float* __restrict__ out,
                        int B, int N, int affine_relu) {
  int f = blockIdx.x * blockDim.x + threadIdx.x;
  if (f >= N) return;
  float s = 0.f, s2 = 0.f;
  for (int b = 0; b < B; ++b) {
    float v = H[(long long)b * N + f];
    s += v; s2 += v * v;
  }
  float mean = s / (float)B;
  float var  = s2 / (float)B - mean * mean;       // biased var (ddof=0)
  float inv  = rsqrtf(var + 1e-5f);
  for (int b = 0; b < B; ++b) {
    float v = (H[(long long)b * N + f] - mean) * inv;
    if (affine_relu) v = fmaxf(v * g[f] + be[f], 0.f);
    out[(long long)b * N + f] = v;
  }
}

// ---------------------------------------------------------------------------
// Host-side orchestration
// ---------------------------------------------------------------------------
static inline int cdiv(long long n, int b) { return (int)((n + b - 1) / b); }

extern "C" void kernel_launch(void* const* d_in, const int* in_sizes, int n_in,
                              void* d_out, int out_size, void* d_ws, size_t ws_size,
                              hipStream_t stream) {
  (void)in_sizes; (void)n_in; (void)out_size; (void)ws_size;
  // setup_inputs() dict order:
  const float* x   = (const float*)d_in[0];                  // [32,16384,3]
  const int*   e[4]  = {(const int*)d_in[1], (const int*)d_in[2],
                        (const int*)d_in[3], (const int*)d_in[4]};
  const int*   l[3]  = {(const int*)d_in[5], (const int*)d_in[6],
                        (const int*)d_in[7]};
  const float* Wc[5] = {(const float*)d_in[8],  (const float*)d_in[10],
                        (const float*)d_in[12], (const float*)d_in[14],
                        (const float*)d_in[16]};
  const float* bc[5] = {(const float*)d_in[9],  (const float*)d_in[11],
                        (const float*)d_in[13], (const float*)d_in[15],
                        (const float*)d_in[17]};
  const float* W6 = (const float*)d_in[18];
  const float* g6 = (const float*)d_in[20];  const float* be6 = (const float*)d_in[21];
  const float* W7 = (const float*)d_in[22];
  const float* g7 = (const float*)d_in[24];  const float* be7 = (const float*)d_in[25];
  const float* W8 = (const float*)d_in[26];
  const float* g8 = (const float*)d_in[28];  const float* be8 = (const float*)d_in[29];
  const float* W9 = (const float*)d_in[30];
  // biases b6..b9 cancel under batch-norm mean subtraction -> not needed.

  const int B = 32, Hc = 32, F = 4096, L = 128;
  const int Ns[4] = {16384, 4096, 1024, 128};
  const int Es[4] = {131072, 32768, 8192, 1024};

  // Workspace layout (floats; all offsets stay 32-byte aligned)
  float* ws = (float*)d_ws;
  long long off = 0;
  float* dis[4];
  for (int i = 0; i < 4; ++i) { dis[i] = ws + off; off += Ns[i]; }
  const long long TXCAP = (long long)B * 4096 * Hc;          // 4,194,304 floats
  float* txA  = ws + off; off += TXCAP;
  float* txB  = ws + off; off += TXCAP;
  float* bufX = ws + off; off += TXCAP;
  float* bufY = ws + off; off += TXCAP;
  float* fcA  = ws + off; off += (long long)B * F;
  float* fcB  = ws + off; off += (long long)B * F;
  float* fcC  = ws + off; off += (long long)B * F;
  _Float16* xh = (_Float16*)(ws + off); off += (long long)B * F / 2;  // f16 plane
  _Float16* xl = (_Float16*)(ws + off); off += (long long)B * F / 2;  // f16 plane

  const int TB = 256;

  // ---- degree / D^-1/2 per level ----
  for (int i = 0; i < 4; ++i) {
    zero_f<<<cdiv(Ns[i], TB), TB, 0, stream>>>(dis[i], Ns[i]);
    deg_acc<<<cdiv(Es[i], TB), TB, 0, stream>>>(e[i], Es[i], dis[i]);
    deg_to_dis<<<cdiv(Ns[i], TB), TB, 0, stream>>>(dis[i], Ns[i]);
  }

  // ---- one Chebyshev conv level ----
  auto cheb = [&](const float* xin, int lvl, int N, int C,
                  const float* W, const float* bias,
                  const int* idx, int Nout, int do_tanh, float* out) {
    long long nTx = (long long)B * N * C;
    zero_f<<<cdiv(nTx, TB), TB, 0, stream>>>(txA, nTx);
    prop_k<<<cdiv((long long)Es[lvl] * C, TB), TB, 0, stream>>>(
        xin, e[lvl], dis[lvl], txA, B, N, C, Es[lvl]);
    zero_f<<<cdiv(nTx, TB), TB, 0, stream>>>(txB, nTx);
    prop_k<<<cdiv((long long)Es[lvl] * C, TB), TB, 0, stream>>>(
        txA, e[lvl], dis[lvl], txB, B, N, C, Es[lvl]);
    long long nOut = (long long)B * Nout * Hc;
    cheb_combine<<<cdiv(nOut, TB), TB, 0, stream>>>(
        xin, txA, txB, W, bias, idx, out, B, N, C, Hc, Nout, do_tanh);
  };

  // level 0: C=3, no tanh, pooled by l0 -> bufX [32,4096,32]
  cheb(x,    0, Ns[0], 3,  Wc[0], bc[0], l[0], Ns[1], 0, bufX);
  // level 1: tanh, pooled by l1 -> bufY [32,1024,32]
  cheb(bufX, 1, Ns[1], Hc, Wc[1], bc[1], l[1], Ns[2], 1, bufY);
  // level 2: tanh, pooled by l2 -> bufX [32,128,32]
  cheb(bufY, 2, Ns[2], Hc, Wc[2], bc[2], l[2], Ns[3], 1, bufX);
  // level 3 (Wc4): tanh, no pool -> bufY [32,128,32]
  cheb(bufX, 3, Ns[3], Hc, Wc[3], bc[3], nullptr, Ns[3], 1, bufY);
  // level 3 (Wc5): no tanh -> bufX [32,128,32] == flat [32,4096]
  cheb(bufY, 3, Ns[3], Hc, Wc[4], bc[4], nullptr, Ns[3], 0, bufX);

  // ---- dense head: split-f16 WMMA GEMMs + batch-stat BN ----
  auto gemm = [&](const float* Xm, const float* Wm, float* Om, int K, int N) {
    long long nA = (long long)B * K;
    split_f16<<<cdiv(nA, TB), TB, 0, stream>>>(Xm, xh, xl, nA);
    int waves = N / 16;                 // N is a multiple of 16 here
    int blocks = (waves + 3) / 4;       // 4 waves (128 threads) per block
    gemm32_wmma<<<blocks, 128, 0, stream>>>(xh, xl, Wm, Om, K, N);
  };

  gemm(bufX, W6, fcA, F, F);
  bn_relu<<<cdiv(F, TB), TB, 0, stream>>>(fcA, g6, be6, fcB, B, F, 1);
  gemm(fcB, W7, fcA, F, F);
  bn_relu<<<cdiv(F, TB), TB, 0, stream>>>(fcA, g7, be7, fcC, B, F, 1);
  gemm(fcC, W8, fcA, F, F);
  bn_relu<<<cdiv(F, TB), TB, 0, stream>>>(fcA, g8, be8, fcB, B, F, 1);
  gemm(fcB, W9, fcA, F, L);
  bn_relu<<<cdiv(L, TB), TB, 0, stream>>>(fcA, nullptr, nullptr,
                                          (float*)d_out, B, L, 0);
}
